// AttnModel_2302102471373
// MI455X (gfx1250) — matrix operand order repaired and checked by reading, compiled, not run on hardware
//
#include <hip/hip_runtime.h>
#include <hip/hip_bf16.h>

#define B_ 2048
#define N_ 200
#define FEAT_ 128
#define D_ 256
#define SCALE_ 0.08838834764831845f   // 1/sqrt(128)

typedef float v8f  __attribute__((ext_vector_type(8)));
typedef __bf16 v16bf __attribute__((ext_vector_type(16)));

// ---------------------------------------------------------------------------
// wave32 shuffle reductions
// ---------------------------------------------------------------------------
__device__ __forceinline__ float wave_sum(float v) {
#pragma unroll
    for (int o = 16; o > 0; o >>= 1) v += __shfl_xor(v, o, 32);
    return v;
}
__device__ __forceinline__ float wave_max(float v) {
#pragma unroll
    for (int o = 16; o > 0; o >>= 1) v = fmaxf(v, __shfl_xor(v, o, 32));
    return v;
}

// ---------------------------------------------------------------------------
// q_in = concat(src, src_t)  -> (B, 256)
// ---------------------------------------------------------------------------
__global__ void qin_concat_kernel(const float* __restrict__ src,
                                  const float* __restrict__ src_t,
                                  float* __restrict__ qin) {
    int b = blockIdx.x, j = threadIdx.x;
    qin[b * D_ + j] = (j < FEAT_) ? src[b * FEAT_ + j] : src_t[b * FEAT_ + (j - FEAT_)];
}

// ---------------------------------------------------------------------------
// LDS-staged WMMA GEMM:  C[m,n] = act( sum_k A[m,k]*B(n,k) + bias[n] )
//   A row-major (lda); B element (n,k) at Bp[n*bNs + k*bKs] (bKs!=1 -> gather).
//   bf16 hi/lo split x v_wmma_f32_16x16x32_bf16 (3 wmma / k-step) ~ fp32 acc.
//   Block = 256 thr (8 waves, 2Mx4N); block tile 32M x 128N; each wave owns
//   two 16x16 accumulators. A/B tiles staged once per block through LDS
//   (float4 global loads, 16B-aligned LDS reads, stride 36 to spread banks).
//   Requires M%32==0, N%128==0, K%32==0.
// ---------------------------------------------------------------------------
__global__ void wmma_gemm_kernel(const float* __restrict__ A, int lda,
                                 const float* __restrict__ Bp, int bNs, int bKs,
                                 const float* __restrict__ bias,
                                 float* __restrict__ C, int ldc,
                                 int K, int relu) {
    __shared__ float As[32 * 36];    //  4.5 KB
    __shared__ float Bs[128 * 36];   // 18.0 KB

    const int tid  = threadIdx.x;
    const int lane = tid & 31;
    const int wave = tid >> 5;
    const int r16  = lane & 15;
    const int half = lane >> 4;          // 0: lanes 0-15, 1: lanes 16-31
    const int h8   = 8 * half;
    const int wm   = wave >> 2;          // 0..1  (M tile within block)
    const int wn   = wave & 3;           // 0..3  (N group within block)
    const int blockM = blockIdx.y * 32;
    const int blockN = blockIdx.x * 128;

    const int ar  = tid >> 3;            // 0..31
    const int ak4 = (tid & 7) * 4;       // 0,4,...,28

    v8f acc0 = {}, acc1 = {};

    for (int k0 = 0; k0 < K; k0 += 32) {
        // ---- cooperative stage: A 32x32, B 128x32 ----
        {
            const float* ap = A + (size_t)(blockM + ar) * lda + k0 + ak4;
            *(float4*)&As[ar * 36 + ak4] = *(const float4*)ap;
        }
        if (bKs == 1) {
#pragma unroll
            for (int i = 0; i < 4; ++i) {
                const int bn = 32 * i + ar;
                const float* bp = Bp + (size_t)(blockN + bn) * bNs + k0 + ak4;
                *(float4*)&Bs[bn * 36 + ak4] = *(const float4*)bp;
            }
        } else {
            // transposed-B gather path
            for (int idx = tid; idx < 128 * 32; idx += 256) {
                const int bn = idx >> 5, bk = idx & 31;
                Bs[bn * 36 + bk] =
                    Bp[(size_t)(blockN + bn) * bNs + (size_t)(k0 + bk) * bKs];
            }
        }
        __syncthreads();

        // ---- A fragment (ISA dense 16-bit A layout), hi/lo split ----
        v16bf ah, al;
        {
            const float* arow = &As[(wm * 16 + r16) * 36];
#pragma unroll
            for (int i = 0; i < 8; ++i) {
                float x0 = arow[h8 + i];
                float x1 = arow[16 + h8 + i];
                __bf16 hx0 = (__bf16)x0, hx1 = (__bf16)x1;
                ah[i]     = hx0;  al[i]     = (__bf16)(x0 - (float)hx0);
                ah[8 + i] = hx1;  al[8 + i] = (__bf16)(x1 - (float)hx1);
            }
        }

        // ---- two B fragments, two accumulators, 3 wmma each ----
#pragma unroll
        for (int t = 0; t < 2; ++t) {
            const float* brow = &Bs[(wn * 32 + t * 16 + r16) * 36];
            v16bf bh, bl;
#pragma unroll
            for (int i = 0; i < 16; ++i) {
                float xb = brow[(i < 8) ? (h8 + i) : (16 + h8 + (i - 8))];
                __bf16 hb = (__bf16)xb;
                bh[i] = hb;  bl[i] = (__bf16)(xb - (float)hb);
            }
            v8f a = (t == 0) ? acc0 : acc1;
            a = __builtin_amdgcn_wmma_f32_16x16x32_bf16(false, al, false, bh, (short)0, a, false, false);
            a = __builtin_amdgcn_wmma_f32_16x16x32_bf16(false, ah, false, bl, (short)0, a, false, false);
            a = __builtin_amdgcn_wmma_f32_16x16x32_bf16(false, ah, false, bh, (short)0, a, false, false);
            if (t == 0) acc0 = a; else acc1 = a;
        }
        __syncthreads();
    }

    // ---- epilogue: C/D f32 layout (lane col = r16, VGPR r -> row r+8*half)
#pragma unroll
    for (int t = 0; t < 2; ++t) {
        const v8f a = (t == 0) ? acc0 : acc1;
        const int col = blockN + wn * 32 + t * 16 + r16;
        const float bv = bias ? bias[col] : 0.0f;
        float* Cb = C + (size_t)(blockM + wm * 16 + 8 * half) * ldc + col;
#pragma unroll
        for (int r = 0; r < 8; ++r) {
            float v = a[r] + bv;
            if (relu) v = fmaxf(v, 0.0f);
            Cb[(size_t)r * ldc] = v;
        }
    }
}

// ---------------------------------------------------------------------------
// Streaming attention (the HBM-bound part): one workgroup (8 waves) per b.
//   scores[h,n] = w[b,h,:] . k_in[b,n,:] * scale  (masked)  -> softmax ->
//   attn_out[b,n] = mean_h attn ;  ctx[b,h,:] = sum_n attn[h,n]*k_in[b,n,:]
// ---------------------------------------------------------------------------
__global__ void attn_stream_kernel(const float* __restrict__ seq,
                                   const float* __restrict__ seq_t,
                                   const float* __restrict__ wv0,  // (B,256)
                                   const float* __restrict__ wv1,  // (B,256)
                                   const unsigned char* __restrict__ mask,
                                   float* __restrict__ ctx0,       // (B,256)
                                   float* __restrict__ ctx1,       // (B,256)
                                   float* __restrict__ attn_out) { // (B,N)
    __shared__ float w0[D_], w1[D_];
    __shared__ float s0[N_], s1[N_];
    __shared__ float sinv[2];

    const int b    = blockIdx.x;
    const int tid  = threadIdx.x;
    const int lane = tid & 31;
    const int wave = tid >> 5;

    w0[tid] = wv0[(size_t)b * D_ + tid];
    w1[tid] = wv1[(size_t)b * D_ + tid];
    __syncthreads();

    const float* seqb  = seq   + (size_t)b * N_ * FEAT_;
    const float* seqtb = seq_t + (size_t)b * N_ * FEAT_;

    // ---- phase 1: scores, one wave per row n (coalesced 128B/wave loads)
    for (int n = wave; n < N_; n += 8) {
        const float* sr = seqb  + (size_t)n * FEAT_;
        const float* st = seqtb + (size_t)n * FEAT_;
        float a0 = 0.f, a1 = 0.f;
#pragma unroll
        for (int j = lane; j < FEAT_; j += 32) {
            float v = sr[j];
            a0 = fmaf(v, w0[j], a0);
            a1 = fmaf(v, w1[j], a1);
        }
#pragma unroll
        for (int j = lane; j < FEAT_; j += 32) {
            float v = st[j];
            a0 = fmaf(v, w0[FEAT_ + j], a0);
            a1 = fmaf(v, w1[FEAT_ + j], a1);
        }
        a0 = wave_sum(a0);
        a1 = wave_sum(a1);
        if (lane == 0) {
            bool mk = mask[(size_t)b * N_ + n] != 0;
            s0[n] = mk ? -1e10f : a0 * SCALE_;
            s1[n] = mk ? -1e10f : a1 * SCALE_;
        }
    }
    __syncthreads();

    // ---- phase 2: softmax per head (wave 0 -> h0, wave 1 -> h1)
    if (wave < 2) {
        float* s = (wave == 0) ? s0 : s1;
        float m = -3.4e38f;
        for (int n = lane; n < N_; n += 32) m = fmaxf(m, s[n]);
        m = wave_max(m);
        float sum = 0.f;
        for (int n = lane; n < N_; n += 32) {
            float e = expf(s[n] - m);
            s[n] = e;
            sum += e;
        }
        sum = wave_sum(sum);
        if (lane == 0) sinv[wave] = 1.0f / sum;
    }
    __syncthreads();

    const float i0 = sinv[0], i1 = sinv[1];
    for (int n = tid; n < N_; n += 256) {
        float a0 = s0[n] * i0, a1 = s1[n] * i1;
        s0[n] = a0;
        s1[n] = a1;
        attn_out[(size_t)b * N_ + n] = 0.5f * (a0 + a1);
    }
    __syncthreads();

    // ---- phase 3: ctx accumulation, thread-per-column (coalesced; L2 hits)
    const int j = tid;
    const float* base = (j < FEAT_) ? (seqb + j) : (seqtb + (j - FEAT_));
    float acc0 = 0.f, acc1 = 0.f;
    for (int n = 0; n < N_; ++n) {
        if (n + 8 < N_) __builtin_prefetch(base + (size_t)(n + 8) * FEAT_, 0, 1);
        float v = base[(size_t)n * FEAT_];
        acc0 = fmaf(s0[n], v, acc0);
        acc1 = fmaf(s1[n], v, acc1);
    }
    ctx0[(size_t)b * D_ + j] = acc0;
    ctx1[(size_t)b * D_ + j] = acc1;
}

// ---------------------------------------------------------------------------
// LayerNorm( y + qin ) -> merged = [ xn , src ]   (B x 384)
// ---------------------------------------------------------------------------
__global__ void ln_merge_kernel(const float* __restrict__ y,
                                const float* __restrict__ qin,
                                const float* __restrict__ src,
                                const float* __restrict__ g,
                                const float* __restrict__ beta,
                                float* __restrict__ merged) {
    __shared__ float ls[8], ls2[8];
    const int b = blockIdx.x, j = threadIdx.x;
    const int lane = j & 31, wave = j >> 5;

    float x = y[(size_t)b * D_ + j] + qin[(size_t)b * D_ + j];
    float s  = wave_sum(x);
    float s2 = wave_sum(x * x);
    if (lane == 0) { ls[wave] = s; ls2[wave] = s2; }
    __syncthreads();
    float tot = 0.f, tot2 = 0.f;
#pragma unroll
    for (int i = 0; i < 8; ++i) { tot += ls[i]; tot2 += ls2[i]; }
    const float mu  = tot * (1.0f / D_);
    const float var = tot2 * (1.0f / D_) - mu * mu;
    const float inv = rsqrtf(var + 1e-5f);

    float xn = (x - mu) * inv * g[j] + beta[j];
    merged[(size_t)b * 384 + j] = xn;
    if (j < FEAT_) merged[(size_t)b * 384 + D_ + j] = src[(size_t)b * FEAT_ + j];
}

// ---------------------------------------------------------------------------
// launcher
// ---------------------------------------------------------------------------
extern "C" void kernel_launch(void* const* d_in, const int* in_sizes, int n_in,
                              void* d_out, int out_size, void* d_ws, size_t ws_size,
                              hipStream_t stream) {
    (void)in_sizes; (void)n_in; (void)out_size; (void)ws_size;

    const float* src   = (const float*)d_in[0];
    const float* src_t = (const float*)d_in[1];
    const float* seq   = (const float*)d_in[2];
    const float* seq_t = (const float*)d_in[3];
    const float* Wq    = (const float*)d_in[4];
    const float* Wk    = (const float*)d_in[5];
    const float* Wv    = (const float*)d_in[6];
    const float* Wfc   = (const float*)d_in[7];
    const float* bfc   = (const float*)d_in[8];
    const float* ln_g  = (const float*)d_in[9];
    const float* ln_b  = (const float*)d_in[10];
    const float* W1    = (const float*)d_in[11];
    const float* b1    = (const float*)d_in[12];
    const float* W2    = (const float*)d_in[13];
    const float* b2    = (const float*)d_in[14];
    const unsigned char* mask = (const unsigned char*)d_in[15];

    float* ws = (float*)d_ws;
    const size_t P = (size_t)B_ * D_;          // 524288 floats
    float* qin    = ws;                        // P               [live until LN]
    float* q      = ws + P;                    // P  (reused as y)
    float* wvec   = ws + 2 * P;                // 2P (reused as merged + h1)
    float* ctx    = ws + 4 * P;                // 2P
    float* outp   = ws + 6 * P;                // P    total 7P floats = 14.7 MB
    float* y      = q;
    float* merged = wvec;                      // B x 384
    float* h1     = wvec + (size_t)B_ * 384;   // B x 128 (fits inside wvec's 2P)
    float* out_final = (float*)d_out;                       // B x 128
    float* attn_out  = (float*)d_out + (size_t)B_ * FEAT_;  // B x 200

    // 1) q_in = [src, src_t]
    qin_concat_kernel<<<B_, 256, 0, stream>>>(src, src_t, qin);

    // 2) q = q_in @ Wq^T          (2048x256 x K=256)
    wmma_gemm_kernel<<<dim3(256 / 128, B_ / 32), 256, 0, stream>>>(
        qin, D_, Wq, D_, 1, nullptr, q, D_, D_, 0);

    // 3) w_h = q_h @ Wk_h  (B^T gather: B(n,k) = Wk[h*128+k, n]), K=128
    for (int h = 0; h < 2; ++h)
        wmma_gemm_kernel<<<dim3(256 / 128, B_ / 32), 256, 0, stream>>>(
            q + h * 128, D_, Wk + (size_t)h * 128 * D_, 1, D_,
            nullptr, wvec + (size_t)h * P, D_, 128, 0);

    // 4) streaming attention: scores/softmax/ctx/attn_out
    attn_stream_kernel<<<B_, 256, 0, stream>>>(
        seq, seq_t, wvec, wvec + P, mask, ctx, ctx + P, attn_out);

    // 5) out_h = ctx_h @ Wv_h^T   (2048x128 x K=256), packed at col offset h*128
    for (int h = 0; h < 2; ++h)
        wmma_gemm_kernel<<<dim3(128 / 128, B_ / 32), 256, 0, stream>>>(
            ctx + (size_t)h * P, D_, Wv + (size_t)h * 128 * D_, D_, 1,
            nullptr, outp + h * 128, D_, D_, 0);

    // 6) y = out @ Wfc^T + bfc    (2048x256 x K=256)
    wmma_gemm_kernel<<<dim3(256 / 128, B_ / 32), 256, 0, stream>>>(
        outp, D_, Wfc, D_, 1, bfc, y, D_, D_, 0);

    // 7) layernorm(y + qin) ; merged = [xn, src]
    ln_merge_kernel<<<B_, 256, 0, stream>>>(y, qin, src, ln_g, ln_b, merged);

    // 8) h1 = relu(merged @ W1^T + b1)   (2048x128 x K=384)
    wmma_gemm_kernel<<<dim3(128 / 128, B_ / 32), 256, 0, stream>>>(
        merged, 384, W1, 384, 1, b1, h1, 128, 384, 1);

    // 9) out_final = h1 @ W2^T + b2      (2048x128 x K=128)
    wmma_gemm_kernel<<<dim3(128 / 128, B_ / 32), 256, 0, stream>>>(
        h1, 128, W2, 128, 1, b2, out_final, 128, 128, 0);
}
